// CombinedPretrainLoss_38439957299272
// MI455X (gfx1250) — matrix operand order repaired and checked by reading, not compiled, not run on hardware
//
#include <hip/hip_runtime.h>
#include <math.h>

// ---------------------------------------------------------------------------
// CombinedPretrainLoss for MI455X (gfx1250, wave32, WMMA)
//
// Heavy part: logsumexp over rows of ([z;g] @ memory_queue^T)/TAU,
// 528x131072x256, done with split-bf16 (hi+lo) v_wmma_f32_16x16x32_bf16
// (3 WMMAs per K-step -> near-fp32 accuracy) + online softmax in registers.
// Round 2: two independent WMMA accumulator chains per tile (ILP across the
// XDL pipe) + global_prefetch_b8 of the next queue tile.
// ---------------------------------------------------------------------------

typedef __bf16 bhalf;
typedef __attribute__((ext_vector_type(16))) __bf16 v16bf;
typedef __attribute__((ext_vector_type(8)))  __bf16 v8bf;
typedef __attribute__((ext_vector_type(8)))  float  v8f;

#define DD      256          // feature dim
#define NFR     512          // B*L frames
#define NROWS   528          // frames + 16 g rows
#define MPAD    640          // padded to 5 m-blocks of 128 rows
#define KQ      131072       // queue rows
#define KCHUNKS 64
#define QCHUNK  (KQ / KCHUNKS)   // 2048 queue rows per workgroup
#define QTILE   32               // queue rows staged in LDS at a time
#define LSTR    264              // LDS row stride in halves (256 + 8 pad)
#define INV_TAU (1.0f / 0.07f)
#define NEGBIG  (-3.0e38f)

// ---- helpers --------------------------------------------------------------

__device__ __forceinline__ bhalf f2bf(float f) {
  unsigned u = __float_as_uint(f);
  unsigned r = (u + 0x7FFFu + ((u >> 16) & 1u)) >> 16;   // round-nearest-even
  unsigned short hv = (unsigned short)r;
  return __builtin_bit_cast(bhalf, hv);
}
__device__ __forceinline__ float bf2f(bhalf h) {
  unsigned short hv = __builtin_bit_cast(unsigned short, h);
  return __uint_as_float(((unsigned)hv) << 16);
}
__device__ __forceinline__ void lse_combine(float& M, float& S, float m, float s) {
  float nm = fmaxf(M, m);
  S = S * __expf(M - nm) + s * __expf(m - nm);
  M = nm;
}
__device__ __forceinline__ float dlogaddexp(float a, float b) {
  float m = fmaxf(a, b);
  return __logf(__expf(a - m) + __expf(b - m)) + m;
}
__device__ __forceinline__ v16bf cat8(v8bf lo, v8bf hi) {
  return __builtin_shufflevector(lo, hi, 0,1,2,3,4,5,6,7,8,9,10,11,12,13,14,15);
}

// ---- kernel 1: build split-bf16 R = [z ; g ; zero-pad] ---------------------

__global__ __launch_bounds__(256) void cvt_split_kernel(
    const float* __restrict__ z, const float* __restrict__ g,
    bhalf* __restrict__ Rhi, bhalf* __restrict__ Rlo)
{
  int idx = blockIdx.x * 256 + threadIdx.x;     // [0, MPAD*DD)
  int row = idx / DD, d = idx - row * DD;
  float v = 0.0f;
  if (row < NFR)        v = z[idx];
  else if (row < NROWS) v = g[(row - NFR) * DD + d];
  bhalf h = f2bf(v);
  Rhi[idx] = h;
  Rlo[idx] = f2bf(v - bf2f(h));
}

// ---- kernel 2: queue logsumexp partials via WMMA ---------------------------
// grid: (KCHUNKS, MPAD/128), 256 threads = 8 waves; wave w owns M-tile
// rows [mblock*128 + w*16, +16).  Online softmax over its queue chunk.

__global__ __launch_bounds__(256) void queue_lse_kernel(
    const float* __restrict__ q,
    const bhalf* __restrict__ Rhi, const bhalf* __restrict__ Rlo,
    float* __restrict__ pmax, float* __restrict__ psum)
{
  __shared__ __attribute__((aligned(16))) bhalf qhi[QTILE * LSTR];
  __shared__ __attribute__((aligned(16))) bhalf qlo[QTILE * LSTR];

  const int tid   = threadIdx.x;
  const int lane  = tid & 31;
  const int wave  = tid >> 5;
  const int chunk = blockIdx.x;
  const int mbase = blockIdx.y * 128 + wave * 16;

  // A fragments (16-bit A 16x32 ISA layout): lane<16 -> M=lane, K {0-7,16-23};
  // lane>=16 -> M=lane-16, K {8-15,24-31}; per K-step of 32.
  const int arow = mbase + (lane & 15);
  const int aofs = (lane < 16) ? 0 : 8;
  v16bf ah[8], al[8];
#pragma unroll
  for (int ks = 0; ks < 8; ++ks) {
    const bhalf* ph = Rhi + arow * DD + ks * 32 + aofs;
    const bhalf* pl = Rlo + arow * DD + ks * 32 + aofs;
    ah[ks] = cat8(*(const v8bf*)(ph), *(const v8bf*)(ph + 16));
    al[ks] = cat8(*(const v8bf*)(pl), *(const v8bf*)(pl + 16));
  }

  float rmax[8], rsum[8];
#pragma unroll
  for (int r = 0; r < 8; ++r) { rmax[r] = NEGBIG; rsum[r] = 0.0f; }

  const int qrow0 = chunk * QCHUNK;
  for (int qt = 0; qt < QCHUNK; qt += QTILE) {
    // stage QTILE queue rows: f32 -> split bf16 in LDS (all 8 waves share)
    for (int i = tid; i < QTILE * (DD / 4); i += 256) {
      int r  = i >> 6;                 // 64 float4 per row
      int c4 = i & 63;
      const float4 v = ((const float4*)(q + (size_t)(qrow0 + qt + r) * DD))[c4];
      bhalf* dh = &qhi[r * LSTR + c4 * 4];
      bhalf* dl = &qlo[r * LSTR + c4 * 4];
      float vv[4] = {v.x, v.y, v.z, v.w};
#pragma unroll
      for (int j = 0; j < 4; ++j) {
        bhalf h = f2bf(vv[j]);
        dh[j] = h;
        dl[j] = f2bf(vv[j] - bf2f(h));
      }
    }
    __syncthreads();

    // prefetch next tile toward cache while this tile's WMMAs run
    // (emits global_prefetch_b8; 256 threads x 128B = 32KB tile)
    if (qt + QTILE < QCHUNK) {
      const float* np = q + (size_t)(qrow0 + qt + QTILE + (tid >> 3)) * DD
                          + (size_t)(tid & 7) * 32;
      __builtin_prefetch(np, 0, 1);
    }

    // Two 16-column subtiles processed as two INDEPENDENT accumulator
    // chains, interleaved -> back-to-back WMMAs without C-operand RAW stalls.
    // B fragment (16-bit B 32x16): lane<16 -> N=lane, K {0-7,16-23};
    // lane>=16 -> N=lane-16, K {8-15,24-31}.
    const int bofs = (lane < 16) ? 0 : 8;
    const bhalf* bh0 = &qhi[(lane & 15) * LSTR + bofs];
    const bhalf* bl0 = &qlo[(lane & 15) * LSTR + bofs];
    const bhalf* bh1 = &qhi[(16 + (lane & 15)) * LSTR + bofs];
    const bhalf* bl1 = &qlo[(16 + (lane & 15)) * LSTR + bofs];
    v8f acc0 = {0.f, 0.f, 0.f, 0.f, 0.f, 0.f, 0.f, 0.f};
    v8f acc1 = {0.f, 0.f, 0.f, 0.f, 0.f, 0.f, 0.f, 0.f};
#pragma unroll
    for (int ks = 0; ks < 8; ++ks) {
      v16bf Bh0 = cat8(*(const v8bf*)(bh0 + ks * 32), *(const v8bf*)(bh0 + ks * 32 + 16));
      v16bf Bl0 = cat8(*(const v8bf*)(bl0 + ks * 32), *(const v8bf*)(bl0 + ks * 32 + 16));
      v16bf Bh1 = cat8(*(const v8bf*)(bh1 + ks * 32), *(const v8bf*)(bh1 + ks * 32 + 16));
      v16bf Bl1 = cat8(*(const v8bf*)(bl1 + ks * 32), *(const v8bf*)(bl1 + ks * 32 + 16));
      // split-bf16: aH*bH + aH*bL + aL*bH  (near-fp32 dot product)
      acc0 = __builtin_amdgcn_wmma_f32_16x16x32_bf16(false, ah[ks], false, Bh0, (short)0, acc0, false, false);
      acc1 = __builtin_amdgcn_wmma_f32_16x16x32_bf16(false, ah[ks], false, Bh1, (short)0, acc1, false, false);
      acc0 = __builtin_amdgcn_wmma_f32_16x16x32_bf16(false, ah[ks], false, Bl0, (short)0, acc0, false, false);
      acc1 = __builtin_amdgcn_wmma_f32_16x16x32_bf16(false, ah[ks], false, Bl1, (short)0, acc1, false, false);
      acc0 = __builtin_amdgcn_wmma_f32_16x16x32_bf16(false, al[ks], false, Bh0, (short)0, acc0, false, false);
      acc1 = __builtin_amdgcn_wmma_f32_16x16x32_bf16(false, al[ks], false, Bh1, (short)0, acc1, false, false);
    }
    // online logsumexp update: acc0[r]/acc1[r] = S[mrow, qcol0/1] (C/D layout)
#pragma unroll
    for (int r = 0; r < 8; ++r) {
      float v0 = acc0[r] * INV_TAU;
      float nm = fmaxf(rmax[r], v0);
      rsum[r] = rsum[r] * __expf(rmax[r] - nm) + __expf(v0 - nm);
      rmax[r] = nm;
      float v1 = acc1[r] * INV_TAU;
      nm = fmaxf(rmax[r], v1);
      rsum[r] = rsum[r] * __expf(rmax[r] - nm) + __expf(v1 - nm);
      rmax[r] = nm;
    }
    __syncthreads();
  }

  // combine across the 16-lane half (each row's N-values live across 16 lanes)
#pragma unroll
  for (int m = 1; m <= 8; m <<= 1) {
#pragma unroll
    for (int r = 0; r < 8; ++r) {
      float om = __shfl_xor(rmax[r], m, 32);
      float os = __shfl_xor(rsum[r], m, 32);
      float nm = fmaxf(rmax[r], om);
      rsum[r] = rsum[r] * __expf(rmax[r] - nm) + os * __expf(om - nm);
      rmax[r] = nm;
    }
  }
  if ((lane & 15) == 0) {
    const int rbase = mbase + ((lane < 16) ? 0 : 8);  // lanes<16: M 0-7, else 8-15
#pragma unroll
    for (int r = 0; r < 8; ++r) {
      pmax[(rbase + r) * KCHUNKS + chunk] = rmax[r];
      psum[(rbase + r) * KCHUNKS + chunk] = rsum[r];
    }
  }
}

// ---- kernel 3: frame-vs-frame similarities (tiny GEMM + masked lse) --------
// one workgroup per row r (0..511 = z rows, 512..527 = g rows)

__global__ __launch_bounds__(256) void frames_kernel(
    const float* __restrict__ z, const float* __restrict__ g,
    float* __restrict__ lseF, float* __restrict__ posLL, float* __restrict__ posGL)
{
  __shared__ float avec[DD];
  __shared__ float redm[256], reds[256];
  const int r   = blockIdx.x;
  const int tid = threadIdx.x;
  avec[tid] = (r < NFR) ? z[r * DD + tid] : g[(r - NFR) * DD + tid];
  __syncthreads();

  float M = NEGBIG, S = 0.0f;
#pragma unroll
  for (int cc = 0; cc < 2; ++cc) {
    const int c = tid + cc * 256;            // column (frame) index
    float dot = 0.0f;
    const float* zc = z + c * DD;
    for (int d = 0; d < DD; ++d) dot = fmaf(avec[d], zc[d], dot);
    const float s = dot * INV_TAU;
    bool masked;
    if (r < NFR) {
      masked = (c == r) || (c == r + 1);
      if ((c == r + 1) && ((r & 31) != 31)) posLL[r] = s;   // positive (b,t+1)
    } else {
      const int b = r - NFR;
      masked = ((c >> 5) == b);                             // whole own clip
      if (masked) posGL[b * 32 + (c & 31)] = s;             // positives per t
    }
    if (!masked) lse_combine(M, S, s, 1.0f);
  }
  redm[tid] = M; reds[tid] = S;
  __syncthreads();
  for (int ofs = 128; ofs > 0; ofs >>= 1) {
    if (tid < ofs) lse_combine(redm[tid], reds[tid], redm[tid + ofs], reds[tid + ofs]);
    __syncthreads();
  }
  if (tid == 0) lseF[r] = __logf(reds[0]) + redm[0];
}

// ---- kernel 4: finalize ----------------------------------------------------

__global__ __launch_bounds__(256) void finalize_kernel(
    const float* __restrict__ z,
    const float* __restrict__ pmax, const float* __restrict__ psum,
    const float* __restrict__ lseF, const float* __restrict__ posLL,
    const float* __restrict__ posGL, float* __restrict__ out)
{
  __shared__ float lseQ[NROWS];
  __shared__ float red[256];
  __shared__ float part[3];
  const int tid = threadIdx.x;

  // merge per-chunk queue partials -> per-row queue logsumexp
  for (int row = tid; row < NROWS; row += 256) {
    float M = NEGBIG, S = 0.0f;
    for (int c = 0; c < KCHUNKS; ++c)
      lse_combine(M, S, pmax[row * KCHUNKS + c], psum[row * KCHUNKS + c]);
    lseQ[row] = __logf(S) + M;
  }
  __syncthreads();

  // loss_ll over 496 anchors
  float local = 0.0f;
  for (int r = tid; r < NFR; r += 256) {
    if ((r & 31) != 31) {
      float ln = dlogaddexp(lseF[r], lseQ[r]);
      float p  = posLL[r];
      local += dlogaddexp(p, ln) - p;
    }
  }
  red[tid] = local; __syncthreads();
  for (int ofs = 128; ofs > 0; ofs >>= 1) {
    if (tid < ofs) red[tid] += red[tid + ofs];
    __syncthreads();
  }
  if (tid == 0) part[0] = red[0] / 496.0f;
  __syncthreads();

  // loss_gl over B*L = 512 (pos per t, shared clip-level negatives)
  local = 0.0f;
  for (int i = tid; i < 512; i += 256) {
    int b = i >> 5;
    float ln = dlogaddexp(lseF[NFR + b], lseQ[NFR + b]);
    float p  = posGL[i];
    local += dlogaddexp(p, ln) - p;
  }
  red[tid] = local; __syncthreads();
  for (int ofs = 128; ofs > 0; ofs >>= 1) {
    if (tid < ofs) red[tid] += red[tid + ofs];
    __syncthreads();
  }
  if (tid == 0) part[1] = red[0] / 512.0f;
  __syncthreads();

  // smoothness: sum over b, t<31, d of (z[b,t+1,d]-z[b,t,d])^2, / (B*(L-1))
  local = 0.0f;
  for (int i = tid; i < 16 * 31 * 256; i += 256) {
    int b   = i / (31 * 256);
    int rem = i - b * 31 * 256;
    int t   = rem >> 8;
    int d   = rem & 255;
    float df = z[(b * 32 + t + 1) * 256 + d] - z[(b * 32 + t) * 256 + d];
    local += df * df;
  }
  red[tid] = local; __syncthreads();
  for (int ofs = 128; ofs > 0; ofs >>= 1) {
    if (tid < ofs) red[tid] += red[tid + ofs];
    __syncthreads();
  }
  if (tid == 0) {
    part[2] = red[0] / (16.0f * 31.0f);
    out[0]  = part[0] + 0.5f * part[1] + 0.1f * part[2];
  }
}

// ---- launcher --------------------------------------------------------------

extern "C" void kernel_launch(void* const* d_in, const int* in_sizes, int n_in,
                              void* d_out, int out_size, void* d_ws, size_t ws_size,
                              hipStream_t stream)
{
  const float* z = (const float*)d_in[0];   // [16,32,256]
  const float* g = (const float*)d_in[1];   // [16,256]
  // d_in[2] = va_values: unused by the effective math
  const float* q = (const float*)d_in[3];   // [131072,256]

  char* ws = (char*)d_ws;                   // ~1 MB total
  bhalf* Rhi  = (bhalf*)ws;  ws += (size_t)MPAD * DD * 2;
  bhalf* Rlo  = (bhalf*)ws;  ws += (size_t)MPAD * DD * 2;
  float* pmax = (float*)ws;  ws += (size_t)MPAD * KCHUNKS * 4;
  float* psum = (float*)ws;  ws += (size_t)MPAD * KCHUNKS * 4;
  float* lseF = (float*)ws;  ws += (size_t)NROWS * 4;
  float* posL = (float*)ws;  ws += (size_t)NFR * 4;
  float* posG = (float*)ws;  ws += (size_t)512 * 4;

  cvt_split_kernel<<<(MPAD * DD) / 256, 256, 0, stream>>>(z, g, Rhi, Rlo);

  dim3 gridB(KCHUNKS, MPAD / 128);
  queue_lse_kernel<<<gridB, 256, 0, stream>>>(q, Rhi, Rlo, pmax, psum);

  frames_kernel<<<NROWS, 256, 0, stream>>>(z, g, lseF, posL, posG);

  finalize_kernel<<<1, 256, 0, stream>>>(z, pmax, psum, lseF, posL, posG,
                                         (float*)d_out);
}
